// MultiHeadAttention_48876727828700
// MI455X (gfx1250) — matrix operand order repaired and checked by reading, not compiled, not run on hardware
//
#include <hip/hip_runtime.h>
#include <hip/hip_bf16.h>
#include <stdint.h>

#define D_MODEL   2048
#define NUM_HEADS 16
#define D_K       128
#define SEQ_LEN   2048
#define BATCH     2
#define MROWS     (BATCH * SEQ_LEN)   // 4096
#define NKSTEPS   (D_MODEL / 32)      // 64

typedef __attribute__((ext_vector_type(16))) __bf16 v16bf;
typedef __attribute__((ext_vector_type(8)))  float  v8f;

union FragAB { v16bf v; uint4 q[2]; };

__device__ __forceinline__ uint16_t f32_to_bf16(float f) {
  __bf16 h = (__bf16)f;                       // RNE hardware convert
  return __builtin_bit_cast(uint16_t, h);
}
__device__ __forceinline__ float bf16_to_f32(uint16_t u) {
  return (float)__builtin_bit_cast(__bf16, u);
}

// Async global -> LDS 16-byte copy (ASYNCcnt-tracked DMA path).
__device__ __forceinline__ void async_load_b128(const uint16_t* gsrc,
                                                uint16_t* lds_dst) {
  uint32_t l = (uint32_t)(uintptr_t)lds_dst;  // addr[31:0] = LDS byte offset
  asm volatile("global_load_async_to_lds_b128 %0, %1, off"
               :: "v"(l), "v"(gsrc) : "memory");
}

// xor-lane value exchange within each 16-lane half, pure VALU (v_permlane16).
__device__ __forceinline__ float pl16(float v, unsigned s0, unsigned s1) {
  unsigned u = __builtin_bit_cast(unsigned, v);
  unsigned r = __builtin_amdgcn_permlane16(u, u, s0, s1, false, false);
  return __builtin_bit_cast(float, r);
}
// max across the 16 lanes of each half (halves hold different tile rows)
__device__ __forceinline__ float rowmax16(float v) {
  v = fmaxf(v, pl16(v, 0x67452301u, 0xEFCDAB89u));  // xor 1
  v = fmaxf(v, pl16(v, 0x45670123u, 0xCDEF89ABu));  // xor 3
  v = fmaxf(v, pl16(v, 0x32107654u, 0xBA98FEDCu));  // xor 4
  v = fmaxf(v, pl16(v, 0xFEDCBA98u, 0x76543210u));  // xor 8
  return v;
}

// ---------------------------------------------------------------------------
// Kernel 1: fp32 -> bf16 conversion
// ---------------------------------------------------------------------------
__global__ void cvt_bf16_kernel(const float* __restrict__ src,
                                uint16_t* __restrict__ dst, int n) {
  int i = blockIdx.x * blockDim.x + threadIdx.x;
  int stride = gridDim.x * blockDim.x;
  for (; i < n; i += stride) dst[i] = f32_to_bf16(src[i]);
}

// ---------------------------------------------------------------------------
// Kernel 2: bf16 GEMM  Y[m,n] = sum_k A[m,k] * W[n,k] + bias[n]
// WG tile 128x256 (8 waves, each 64x64 = 4x4 WMMA accums).
// K-loop double-buffered through LDS via async global->LDS b128 copies.
// LDS rows padded to 40 elems (80B) -> conflict-free ds_load_b128 frags.
// ---------------------------------------------------------------------------
template <bool F32OUT>
__global__ __launch_bounds__(256) void gemm_kernel(
    const uint16_t* __restrict__ A,
    const uint16_t* __restrict__ W,
    const float* __restrict__ bias,
    uint16_t* __restrict__ outb,
    float* __restrict__ outf) {
  __shared__ __attribute__((aligned(16))) uint16_t Asmem[2][128 * 40];
  __shared__ __attribute__((aligned(16))) uint16_t Wsmem[2][256 * 40];

  const int tid  = threadIdx.x;
  const int lane = tid & 31;
  const int wave = tid >> 5;
  const int wm   = wave & 1;       // 2 waves along M
  const int wn   = wave >> 1;      // 4 waves along N
  const int row0 = blockIdx.x * 128;
  const int col0 = blockIdx.y * 256;
  const int half = lane >> 4;
  const int l16  = lane & 15;

  v8f acc[4][4];
#pragma unroll
  for (int i = 0; i < 4; i++)
#pragma unroll
    for (int j = 0; j < 4; j++)
      acc[i][j] = (v8f){0.f, 0.f, 0.f, 0.f, 0.f, 0.f, 0.f, 0.f};

  auto stage = [&](int buf, int k0) {
    // A tile: 128 rows x 32 elems = 512 x 16B chunks, 2 per thread
#pragma unroll
    for (int c = 0; c < 2; c++) {
      const int id = tid + c * 256;
      const int r = id >> 2, o = (id & 3) * 8;
      async_load_b128(A + (size_t)(row0 + r) * D_MODEL + k0 + o,
                      &Asmem[buf][r * 40 + o]);
    }
    // W tile: 256 rows x 32 elems = 1024 x 16B chunks, 4 per thread
#pragma unroll
    for (int c = 0; c < 4; c++) {
      const int id = tid + c * 256;
      const int r = id >> 2, o = (id & 3) * 8;
      async_load_b128(W + (size_t)(col0 + r) * D_MODEL + k0 + o,
                      &Wsmem[buf][r * 40 + o]);
    }
  };

  stage(0, 0);
  for (int kt = 0; kt < NKSTEPS; kt++) {
    const int buf = kt & 1;
    if (kt + 1 < NKSTEPS) {
      stage(buf ^ 1, (kt + 1) * 32);
      // 6 newest (next-stage) async ops may remain outstanding
      asm volatile("s_wait_asynccnt 0x6" ::: "memory");
    } else {
      asm volatile("s_wait_asynccnt 0x0" ::: "memory");
    }
    __syncthreads();

    FragAB af[4], bfr[4];
#pragma unroll
    for (int i = 0; i < 4; i++) {
      // A-frag: row M=l16; K = {0..7,16..23} (half0) / {8..15,24..31} (half1)
      const uint16_t* p =
          &Asmem[buf][(wm * 64 + 16 * i + l16) * 40 + (half ? 8 : 0)];
      af[i].q[0] = *(const uint4*)(p);
      af[i].q[1] = *(const uint4*)(p + 16);
    }
#pragma unroll
    for (int j = 0; j < 4; j++) {
      // B-frag: col N=l16; K = {0..7,16..23} (half0) / {8..15,24..31} (half1)
      const uint16_t* p =
          &Wsmem[buf][(wn * 64 + 16 * j + l16) * 40 + (half ? 8 : 0)];
      bfr[j].q[0] = *(const uint4*)(p);
      bfr[j].q[1] = *(const uint4*)(p + 16);
    }
#pragma unroll
    for (int i = 0; i < 4; i++)
#pragma unroll
      for (int j = 0; j < 4; j++)
        acc[i][j] = __builtin_amdgcn_wmma_f32_16x16x32_bf16(
            false, af[i].v, false, bfr[j].v, (short)0, acc[i][j], false, false);
    __syncthreads();
  }

#pragma unroll
  for (int i = 0; i < 4; i++) {
#pragma unroll
    for (int j = 0; j < 4; j++) {
      const int colb = col0 + wn * 64 + 16 * j + l16;
      const float bval = bias[colb];
#pragma unroll
      for (int r = 0; r < 8; r++) {
        const int row = row0 + wm * 64 + 16 * i + r + 8 * half;
        const float v = acc[i][j][r] + bval;
        if (F32OUT)
          outf[(size_t)row * D_MODEL + colb] = v;
        else
          outb[(size_t)row * D_MODEL + colb] = f32_to_bf16(v);
      }
    }
  }
}

// ---------------------------------------------------------------------------
// Kernel 3: RoPE on Q/K + repack to [B,H,S,Dk]; V transposed to [B,H,Dk,S].
// 1/sqrt(Dk) is folded into Q here so attention skips score scaling.
// ---------------------------------------------------------------------------
__global__ void rope_pack_kernel(const uint16_t* __restrict__ qp,
                                 const uint16_t* __restrict__ kp,
                                 const uint16_t* __restrict__ vp,
                                 uint16_t* __restrict__ Qb,
                                 uint16_t* __restrict__ Kb,
                                 uint16_t* __restrict__ Vt) {
  size_t idx = (size_t)blockIdx.x * blockDim.x + threadIdx.x;
  const size_t total = (size_t)BATCH * SEQ_LEN * NUM_HEADS * (D_K / 2);
  if (idx >= total) return;
  const int i = (int)(idx % (D_K / 2));
  size_t t = idx / (D_K / 2);
  const int h = (int)(t % NUM_HEADS); t /= NUM_HEADS;
  const int s = (int)(t % SEQ_LEN);
  const int b = (int)(t / SEQ_LEN);

  const float inv_freq = __powf(10000.f, -(float)i / (float)(D_K / 2));
  const float ang = (float)s * inv_freq;
  const float cs = __cosf(ang), sn = __sinf(ang);
  const float qscale = 0.08838834764831845f;  // 1/sqrt(128)

  const size_t src = ((size_t)(b * SEQ_LEN + s)) * D_MODEL + h * D_K + 2 * i;
  const size_t dst =
      ((((size_t)b * NUM_HEADS + h) * SEQ_LEN) + s) * D_K + 2 * i;

  const float qe = bf16_to_f32(qp[src]), qo = bf16_to_f32(qp[src + 1]);
  Qb[dst]     = f32_to_bf16((qe * cs - qo * sn) * qscale);
  Qb[dst + 1] = f32_to_bf16((qe * sn + qo * cs) * qscale);
  const float ke = bf16_to_f32(kp[src]), ko = bf16_to_f32(kp[src + 1]);
  Kb[dst]     = f32_to_bf16(ke * cs - ko * sn);
  Kb[dst + 1] = f32_to_bf16(ke * sn + ko * cs);

  const size_t vdst =
      (((size_t)b * NUM_HEADS + h) * D_K + 2 * i) * SEQ_LEN + s;
  Vt[vdst]           = vp[src];
  Vt[vdst + SEQ_LEN] = vp[src + 1];
}

// ---------------------------------------------------------------------------
// Kernel 4: causal flash attention.
// One wave owns 16 queries; key blocks of 32; online softmax.
// Row max: v_permlane16 (VALU). Row sum: P x ones via WMMA (result is the
// row sum replicated across all lanes -> no cross-lane sum needed).
// The causal mask is applied ONLY on the single diagonal block.
// ---------------------------------------------------------------------------
template <bool MASKED>
__device__ __forceinline__ void attn_block(
    int k0, int q0, int lane,
    const uint16_t* __restrict__ Kbase,
    const uint16_t* __restrict__ Vbase,
    const FragAB (&qf)[4], const FragAB& ones, uint16_t* __restrict__ pl,
    v8f (&O)[8], float (&m)[8], float (&dsum)[8]) {
  const int half = lane >> 4;
  const int l16  = lane & 15;

  v8f s0 = (v8f){0.f, 0.f, 0.f, 0.f, 0.f, 0.f, 0.f, 0.f};
  v8f s1 = s0;
#pragma unroll
  for (int c = 0; c < 4; c++) {
    FragAB kf0, kf1;
    const uint16_t* p0 =
        Kbase + (size_t)(k0 + l16) * D_K + 32 * c + (half ? 8 : 0);
    kf0.q[0] = *(const uint4*)(p0);
    kf0.q[1] = *(const uint4*)(p0 + 16);
    const uint16_t* p1 = p0 + 16 * D_K;
    kf1.q[0] = *(const uint4*)(p1);
    kf1.q[1] = *(const uint4*)(p1 + 16);
    s0 = __builtin_amdgcn_wmma_f32_16x16x32_bf16(
        false, qf[c].v, false, kf0.v, (short)0, s0, false, false);
    s1 = __builtin_amdgcn_wmma_f32_16x16x32_bf16(
        false, qf[c].v, false, kf1.v, (short)0, s1, false, false);
  }

  const int key0 = k0 + l16, key1 = key0 + 16;
  float alpha[8];
#pragma unroll
  for (int r = 0; r < 8; r++) {
    float v0 = s0[r], v1 = s1[r];
    if (MASKED) {
      const int q = q0 + r + 8 * half;
      v0 = (key0 <= q) ? v0 : -1e30f;
      v1 = (key1 <= q) ? v1 : -1e30f;
    }
    const float mnew = fmaxf(m[r], rowmax16(fmaxf(v0, v1)));
    alpha[r] = __expf(m[r] - mnew);
    m[r] = mnew;
    const float p0v = __expf(v0 - mnew);
    const float p1v = __expf(v1 - mnew);
#pragma unroll
    for (int j = 0; j < 8; j++) O[j][r] *= alpha[r];
    // spill P (C-layout) into per-wave LDS tile as bf16
    const int row = r + 8 * half;
    pl[row * 32 + l16]      = f32_to_bf16(p0v);
    pl[row * 32 + 16 + l16] = f32_to_bf16(p1v);
  }

  asm volatile("s_wait_dscnt 0x0" ::: "memory");

  FragAB pf;  // reload P in A-layout (16x32)
  {
    const uint16_t* p = &pl[l16 * 32 + (half ? 8 : 0)];
    pf.q[0] = *(const uint4*)(p);
    pf.q[1] = *(const uint4*)(p + 16);
  }

  // row sums of P via WMMA against all-ones B: every lane gets its row's sum
  v8f rsv = (v8f){0.f, 0.f, 0.f, 0.f, 0.f, 0.f, 0.f, 0.f};
  rsv = __builtin_amdgcn_wmma_f32_16x16x32_bf16(
      false, pf.v, false, ones.v, (short)0, rsv, false, false);
#pragma unroll
  for (int r = 0; r < 8; r++) dsum[r] = dsum[r] * alpha[r] + rsv[r];

#pragma unroll
  for (int j = 0; j < 8; j++) {
    FragAB vf;  // B-frag from transposed V: contiguous per lane
    const uint16_t* p =
        Vbase + (size_t)(16 * j + l16) * SEQ_LEN + k0 + (half ? 8 : 0);
    vf.q[0] = *(const uint4*)(p);
    vf.q[1] = *(const uint4*)(p + 16);
    O[j] = __builtin_amdgcn_wmma_f32_16x16x32_bf16(
        false, pf.v, false, vf.v, (short)0, O[j], false, false);
  }
}

__global__ __launch_bounds__(256) void attn_kernel(
    const uint16_t* __restrict__ Qb,   // [B,H,S,Dk], pre-scaled by 1/sqrt(Dk)
    const uint16_t* __restrict__ Kb,   // [B,H,S,Dk]
    const uint16_t* __restrict__ Vt,   // [B,H,Dk,S]
    uint16_t* __restrict__ attb) {     // [B,S,D_MODEL]
  __shared__ __attribute__((aligned(16))) uint16_t plds[8][16 * 32];

  const int lane = threadIdx.x & 31;
  const int wave = threadIdx.x >> 5;
  const int half = lane >> 4;
  const int l16  = lane & 15;
  const int bh   = blockIdx.y;
  const int b    = bh / NUM_HEADS;
  const int h    = bh % NUM_HEADS;
  const int q0   = blockIdx.x * 128 + wave * 16;

  const uint16_t* Qbase = Qb + (size_t)bh * SEQ_LEN * D_K;
  const uint16_t* Kbase = Kb + (size_t)bh * SEQ_LEN * D_K;
  const uint16_t* Vbase = Vt + (size_t)bh * D_K * SEQ_LEN;

  FragAB qf[4];
#pragma unroll
  for (int c = 0; c < 4; c++) {
    const uint16_t* p =
        Qbase + (size_t)(q0 + l16) * D_K + 32 * c + (half ? 8 : 0);
    qf[c].q[0] = *(const uint4*)(p);
    qf[c].q[1] = *(const uint4*)(p + 16);
  }

  FragAB ones;  // all-ones bf16 B matrix (0x3F80 per element)
  {
    const uint4 ou = {0x3F803F80u, 0x3F803F80u, 0x3F803F80u, 0x3F803F80u};
    ones.q[0] = ou;
    ones.q[1] = ou;
  }

  v8f O[8];
#pragma unroll
  for (int j = 0; j < 8; j++)
    O[j] = (v8f){0.f, 0.f, 0.f, 0.f, 0.f, 0.f, 0.f, 0.f};
  float m[8], dsum[8];
#pragma unroll
  for (int r = 0; r < 8; r++) { m[r] = -1e30f; dsum[r] = 0.f; }

  // full (unmasked) key blocks, then exactly one masked diagonal block
  const int kdiag = (q0 >> 5) << 5;
  for (int k0 = 0; k0 < kdiag; k0 += 32)
    attn_block<false>(k0, q0, lane, Kbase, Vbase, qf, ones, &plds[wave][0],
                      O, m, dsum);
  attn_block<true>(kdiag, q0, lane, Kbase, Vbase, qf, ones, &plds[wave][0],
                   O, m, dsum);

  uint16_t* outp = attb + (size_t)b * SEQ_LEN * D_MODEL + (size_t)h * D_K;
#pragma unroll
  for (int r = 0; r < 8; r++) {
    const float inv = 1.0f / dsum[r];
    const int s = q0 + r + 8 * half;
#pragma unroll
    for (int j = 0; j < 8; j++)
      outp[(size_t)s * D_MODEL + 16 * j + l16] = f32_to_bf16(O[j][r] * inv);
  }
}

// ---------------------------------------------------------------------------
// Host-side launcher
// ---------------------------------------------------------------------------
extern "C" void kernel_launch(void* const* d_in, const int* in_sizes, int n_in,
                              void* d_out, int out_size, void* d_ws,
                              size_t ws_size, hipStream_t stream) {
  const float* x  = (const float*)d_in[0];
  const float* Wq = (const float*)d_in[1];
  const float* bq = (const float*)d_in[2];
  const float* Wk = (const float*)d_in[3];
  const float* bk = (const float*)d_in[4];
  const float* Wv = (const float*)d_in[5];
  const float* bv = (const float*)d_in[6];
  const float* Wo = (const float*)d_in[7];
  const float* bo = (const float*)d_in[8];
  float* out = (float*)d_out;

  const int N_X = MROWS * D_MODEL;      // 8,388,608
  const int N_W = D_MODEL * D_MODEL;    // 4,194,304

  uint16_t* p  = (uint16_t*)d_ws;
  uint16_t* xb   = p; p += N_X;
  uint16_t* wqb  = p; p += N_W;
  uint16_t* wkb  = p; p += N_W;
  uint16_t* wvb  = p; p += N_W;
  uint16_t* wob  = p; p += N_W;
  uint16_t* qp   = p; p += N_X;
  uint16_t* kp   = p; p += N_X;
  uint16_t* vp   = p; p += N_X;
  uint16_t* Qbuf = p; p += N_X;
  uint16_t* Kbuf = p; p += N_X;
  uint16_t* Vtb  = p; p += N_X;
  uint16_t* attb = xb;  // x is dead after projections; reuse its space

  // 1) downconvert activations + weights
  cvt_bf16_kernel<<<N_X / 256, 256, 0, stream>>>(x, xb, N_X);
  cvt_bf16_kernel<<<N_W / 256, 256, 0, stream>>>(Wq, wqb, N_W);
  cvt_bf16_kernel<<<N_W / 256, 256, 0, stream>>>(Wk, wkb, N_W);
  cvt_bf16_kernel<<<N_W / 256, 256, 0, stream>>>(Wv, wvb, N_W);
  cvt_bf16_kernel<<<N_W / 256, 256, 0, stream>>>(Wo, wob, N_W);

  // 2) Q/K/V projections (bf16 out, bias fused)
  dim3 ggrid(MROWS / 128, D_MODEL / 256);
  gemm_kernel<false><<<ggrid, 256, 0, stream>>>(xb, wqb, bq, qp, nullptr);
  gemm_kernel<false><<<ggrid, 256, 0, stream>>>(xb, wkb, bk, kp, nullptr);
  gemm_kernel<false><<<ggrid, 256, 0, stream>>>(xb, wvb, bv, vp, nullptr);

  // 3) RoPE + layout repack (Q pre-scaled by 1/sqrt(Dk))
  {
    const size_t total = (size_t)BATCH * SEQ_LEN * NUM_HEADS * (D_K / 2);
    rope_pack_kernel<<<(unsigned)((total + 255) / 256), 256, 0, stream>>>(
        qp, kp, vp, Qbuf, Kbuf, Vtb);
  }

  // 4) causal flash attention
  attn_kernel<<<dim3(SEQ_LEN / 128, BATCH * NUM_HEADS), 256, 0, stream>>>(
      Qbuf, Kbuf, Vtb, attb);

  // 5) output projection (fp32 out, bias fused)
  gemm_kernel<true><<<ggrid, 256, 0, stream>>>(attb, wob, bo, nullptr, out);
}
